// Transformer_withknowledge_CATsentimentatt2_57148834840716
// MI455X (gfx1250) — compile-verified, matrix operand order checked by reading, not run on hardware
//
#include <hip/hip_runtime.h>
#include <hip/hip_bf16.h>
#include <math.h>

typedef __attribute__((ext_vector_type(16))) __bf16 v16bf;
typedef __attribute__((ext_vector_type(8)))  __bf16 v8bf;
typedef __attribute__((ext_vector_type(8)))  float  v8f;

constexpr int kH  = 768;       // hidden
constexpr int kS  = 512;       // seq len
constexpr int kB  = 32;        // batch
constexpr int kM  = kB * kS;   // 16384 token rows
constexpr int kN1 = 512;       // MLP mid
constexpr int kNA = 100;       // attention proj
constexpr int kNApad = 112;    // 7 * 16

// ---- WMMA fragment loaders (CDNA5 16x16x32 bf16 layout) ----------------
// A (16xK, M=lane&15) and B (KxN, N=lane&15) fragments share the same
// per-lane K mapping: elems 0..7 -> K = kb..kb+7, elems 8..15 -> K = kb+16..kb+23,
// where kb = (lane>>4)*8.

__device__ __forceinline__ v16bf frag_from_f32row(const float* __restrict__ rowptr, int kb) {
  const float4* q0 = (const float4*)(rowptr + kb);
  const float4* q1 = (const float4*)(rowptr + kb + 16);
  float4 a0 = q0[0], a1 = q0[1], b0 = q1[0], b1 = q1[1];
  v16bf f;
  f[0]=(__bf16)a0.x;  f[1]=(__bf16)a0.y;  f[2]=(__bf16)a0.z;  f[3]=(__bf16)a0.w;
  f[4]=(__bf16)a1.x;  f[5]=(__bf16)a1.y;  f[6]=(__bf16)a1.z;  f[7]=(__bf16)a1.w;
  f[8]=(__bf16)b0.x;  f[9]=(__bf16)b0.y;  f[10]=(__bf16)b0.z; f[11]=(__bf16)b0.w;
  f[12]=(__bf16)b1.x; f[13]=(__bf16)b1.y; f[14]=(__bf16)b1.z; f[15]=(__bf16)b1.w;
  return f;
}

__device__ __forceinline__ v16bf frag_from_bf16row(const __bf16* __restrict__ rowptr, int kb) {
  v8bf lo = *(const v8bf*)(rowptr + kb);        // 16B -> global_load_b128
  v8bf hi = *(const v8bf*)(rowptr + kb + 16);
  v16bf f;
#pragma unroll
  for (int e = 0; e < 8; ++e) { f[e] = lo[e]; f[8 + e] = hi[e]; }
  return f;
}

// ---- Weight pre-conversion kernels -------------------------------------
__global__ __launch_bounds__(256) void k_cvt_bf16(
    const float* __restrict__ src, __bf16* __restrict__ dst, int n4)
{
  int i = blockIdx.x * 256 + threadIdx.x;    // i indexes groups of 4
  if (i >= n4) return;
  float4 v = ((const float4*)src)[i];
  __bf16* d = dst + 4 * (size_t)i;
  d[0] = (__bf16)v.x; d[1] = (__bf16)v.y; d[2] = (__bf16)v.z; d[3] = (__bf16)v.w;
}

// Wa1 (100 x 768) -> bf16 padded to 112 x 768 (rows 100..111 zero)
__global__ __launch_bounds__(256) void k_cvt_wa1(
    const float* __restrict__ src, __bf16* __restrict__ dst)
{
  int i = blockIdx.x * 256 + threadIdx.x;    // over 112*768 elements
  if (i >= kNApad * kH) return;
  int row = i / kH;
  dst[i] = (row < kNA) ? (__bf16)src[i] : (__bf16)0.0f;
}

// ---- Kernel 1: fused gather + MLP layer 1, 64x64 wave tiles ------------
// x[m][n] = relu( table[ids[m]] . W1[n] + b1[n] )
__global__ __launch_bounds__(128) void k_gemm1(
    const float* __restrict__ table, const int* __restrict__ ids,
    const __bf16* __restrict__ W1bf, const float* __restrict__ b1,
    __bf16* __restrict__ xws)
{
  int lane  = threadIdx.x & 31;
  int wave  = blockIdx.x * (blockDim.x >> 5) + (threadIdx.x >> 5);
  int mtile = wave >> 3;            // 256 tiles of 64 M-rows
  int ntile = wave & 7;             // 8 tiles of 64 N-cols
  int kb  = (lane >> 4) * 8;
  int l15 = lane & 15;

  const float*  arow[4];
  const __bf16* brow[4];
#pragma unroll
  for (int i = 0; i < 4; ++i)
    arow[i] = table + (size_t)ids[mtile * 64 + i * 16 + l15] * kH;  // gathered A rows
#pragma unroll
  for (int j = 0; j < 4; ++j)
    brow[j] = W1bf + (size_t)(ntile * 64 + j * 16 + l15) * kH;

  v8f acc[4][4] = {};
#pragma unroll 1
  for (int k = 0; k < kH; k += 32) {
    v16bf a[4], b[4];
#pragma unroll
    for (int i = 0; i < 4; ++i) a[i] = frag_from_f32row(arow[i] + k, kb);
#pragma unroll
    for (int j = 0; j < 4; ++j) b[j] = frag_from_bf16row(brow[j] + k, kb);
#pragma unroll
    for (int i = 0; i < 4; ++i)
#pragma unroll
      for (int j = 0; j < 4; ++j)
        acc[i][j] = __builtin_amdgcn_wmma_f32_16x16x32_bf16(
            false, a[i], false, b[j], (short)0, acc[i][j], false, false);
  }

  int mb = (lane >> 4) * 8;
#pragma unroll
  for (int j = 0; j < 4; ++j) {
    int n    = ntile * 64 + j * 16 + l15;
    float bv = b1[n];
#pragma unroll
    for (int i = 0; i < 4; ++i) {
#pragma unroll
      for (int r = 0; r < 8; ++r) {
        int m = mtile * 64 + i * 16 + mb + r;
        xws[(size_t)m * kN1 + n] = (__bf16)fmaxf(acc[i][j][r] + bv, 0.0f);
      }
    }
  }
}

// ---- Kernel 2: MLP layer 2 + mask, 64x64 wave tiles --------------------
// wh[m][n] = mask(m) * relu( x[m] . W2[n] + b2[n] );  writes f32 and bf16 copies
__global__ __launch_bounds__(128) void k_gemm2(
    const __bf16* __restrict__ x, const __bf16* __restrict__ W2bf,
    const float* __restrict__ b2, const int* __restrict__ ids,
    const unsigned char* __restrict__ in_tab,
    float* __restrict__ wh, __bf16* __restrict__ whbf)
{
  int lane  = threadIdx.x & 31;
  int wave  = blockIdx.x * (blockDim.x >> 5) + (threadIdx.x >> 5);
  int mtile = wave / 12;            // 256 tiles of 64 M-rows
  int ntile = wave % 12;            // 12 tiles of 64 N-cols (768)
  int kb  = (lane >> 4) * 8;
  int l15 = lane & 15;

  const __bf16* arow[4];
  const __bf16* brow[4];
#pragma unroll
  for (int i = 0; i < 4; ++i)
    arow[i] = x + (size_t)(mtile * 64 + i * 16 + l15) * kN1;
#pragma unroll
  for (int j = 0; j < 4; ++j)
    brow[j] = W2bf + (size_t)(ntile * 64 + j * 16 + l15) * kN1;

  v8f acc[4][4] = {};
#pragma unroll 1
  for (int k = 0; k < kN1; k += 32) {
    v16bf a[4], b[4];
#pragma unroll
    for (int i = 0; i < 4; ++i) a[i] = frag_from_bf16row(arow[i] + k, kb);
#pragma unroll
    for (int j = 0; j < 4; ++j) b[j] = frag_from_bf16row(brow[j] + k, kb);
#pragma unroll
    for (int i = 0; i < 4; ++i)
#pragma unroll
      for (int j = 0; j < 4; ++j)
        acc[i][j] = __builtin_amdgcn_wmma_f32_16x16x32_bf16(
            false, a[i], false, b[j], (short)0, acc[i][j], false, false);
  }

  int mb = (lane >> 4) * 8;
#pragma unroll
  for (int j = 0; j < 4; ++j) {
    int n    = ntile * 64 + j * 16 + l15;
    float bv = b2[n];
#pragma unroll
    for (int i = 0; i < 4; ++i) {
#pragma unroll
      for (int r = 0; r < 8; ++r) {
        int m = mtile * 64 + i * 16 + mb + r;
        int s = m & (kS - 1);
        bool ok = (s > 0) && (in_tab[ids[m]] != 0);
        float v = ok ? fmaxf(acc[i][j][r] + bv, 0.0f) : 0.0f;
        wh[(size_t)m * kH + n]   = v;
        whbf[(size_t)m * kH + n] = (__bf16)v;
      }
    }
  }
}

// ---- Kernel 3: attention scores ----------------------------------------
// logit[m] = sum_n Wa2[n] * tanh( wh[m] . Wa1[n] + ba1[n] ) + ba2
__global__ __launch_bounds__(128) void k_attn_scores(
    const __bf16* __restrict__ whbf, const __bf16* __restrict__ Wa1bf,
    const float* __restrict__ ba1, const float* __restrict__ Wa2,
    const float* __restrict__ ba2, float* __restrict__ logits)
{
  int lane  = threadIdx.x & 31;
  int mtile = blockIdx.x * (blockDim.x >> 5) + (threadIdx.x >> 5);
  int kb = (lane >> 4) * 8;
  int nrow = lane & 15;

  const __bf16* arow = whbf + (size_t)(mtile * 16 + nrow) * kH;

  v8f acc[7] = {};
#pragma unroll 1
  for (int k = 0; k < kH; k += 32) {
    v16bf a = frag_from_bf16row(arow + k, kb);
#pragma unroll
    for (int t = 0; t < 7; ++t) {
      v16bf b = frag_from_bf16row(Wa1bf + (size_t)(t * 16 + nrow) * kH + k, kb);
      acc[t] = __builtin_amdgcn_wmma_f32_16x16x32_bf16(false, a, false, b, (short)0, acc[t], false, false);
    }
  }

  int n  = lane & 15;
  int mb = (lane >> 4) * 8;
  float sums[8];
#pragma unroll
  for (int r = 0; r < 8; ++r) sums[r] = 0.0f;
#pragma unroll
  for (int t = 0; t < 7; ++t) {
    int ncol = t * 16 + n;
    float w2 = (ncol < kNA) ? Wa2[ncol] : 0.0f;
    float bb = (ncol < kNA) ? ba1[ncol] : 0.0f;
#pragma unroll
    for (int r = 0; r < 8; ++r) sums[r] += w2 * tanhf(acc[t][r] + bb);
  }
#pragma unroll
  for (int r = 0; r < 8; ++r) {
    float v = sums[r];
    v += __shfl_xor(v, 1, 16);
    v += __shfl_xor(v, 2, 16);
    v += __shfl_xor(v, 4, 16);
    v += __shfl_xor(v, 8, 16);
    if ((lane & 15) == 0) logits[mtile * 16 + mb + r] = v + ba2[0];
  }
}

// ---- Kernel 4: softmax (s>0), pooled sum, classifier -------------------
__global__ __launch_bounds__(256) void k_finalize(
    const float* __restrict__ logits, const float* __restrict__ wh,
    const float* __restrict__ hiddens, const float* __restrict__ Wl,
    const float* __restrict__ bl, float* __restrict__ out)
{
  __shared__ float s_attn[kS];
  __shared__ float s_red[256];
  __shared__ float s_sent[kH];

  int b = blockIdx.x;
  int t = threadIdx.x;
  const float* lg = logits + (size_t)b * kS;

  float l0 = (t >= 1) ? lg[t] : -INFINITY;   // s = t          (s=0 masked out)
  float l1 = lg[t + 256];                    // s = t+256 .. 511

  // max reduce
  s_red[t] = fmaxf(l0, l1);
  __syncthreads();
  for (int off = 128; off > 0; off >>= 1) {
    if (t < off) s_red[t] = fmaxf(s_red[t], s_red[t + off]);
    __syncthreads();
  }
  float mx = s_red[0];
  __syncthreads();

  float e0 = (t >= 1) ? expf(l0 - mx) : 0.0f;
  float e1 = expf(l1 - mx);
  s_attn[t] = e0;
  s_attn[t + 256] = e1;
  s_red[t] = e0 + e1;
  __syncthreads();
  for (int off = 128; off > 0; off >>= 1) {
    if (t < off) s_red[t] += s_red[t + off];
    __syncthreads();
  }
  float inv = 1.0f / s_red[0];
  __syncthreads();

  // sent[h] = sum_s attn[s] * wh[b,s,h]  (h split 3-way over 256 threads)
  float a0 = 0.0f, a1 = 0.0f, a2 = 0.0f;
  const float* whb = wh + (size_t)b * kS * kH;
  for (int s = 1; s < kS; ++s) {
    float w = s_attn[s];
    const float* row = whb + (size_t)s * kH;
    a0 += w * row[t];
    a1 += w * row[t + 256];
    a2 += w * row[t + 512];
  }
  s_sent[t]       = a0 * inv;
  s_sent[t + 256] = a1 * inv;
  s_sent[t + 512] = a2 * inv;
  __syncthreads();

  // predicts = [hiddens[b,0,:], sent] @ Wl^T + bl
  const float* hid0 = hiddens + (size_t)b * kS * kH;  // hiddens[b,0,:]
  for (int o = 0; o < 3; ++o) {
    float p = 0.0f;
#pragma unroll
    for (int j = 0; j < 3; ++j) {
      int h = t + 256 * j;
      p += Wl[o * 2 * kH + h] * hid0[h] + Wl[o * 2 * kH + kH + h] * s_sent[h];
    }
    s_red[t] = p;
    __syncthreads();
    for (int off = 128; off > 0; off >>= 1) {
      if (t < off) s_red[t] += s_red[t + off];
      __syncthreads();
    }
    if (t == 0) out[b * 3 + o] = s_red[0] + bl[o];
    __syncthreads();
  }
}

extern "C" void kernel_launch(void* const* d_in, const int* in_sizes, int n_in,
                              void* d_out, int out_size, void* d_ws, size_t ws_size,
                              hipStream_t stream) {
  (void)in_sizes; (void)n_in; (void)out_size; (void)ws_size;
  const float*         hiddens = (const float*)d_in[0];
  const int*           ids     = (const int*)d_in[1];
  const unsigned char* in_tab  = (const unsigned char*)d_in[2];  // jax bool = 1 byte
  const float*         table   = (const float*)d_in[3];
  const float*         W1      = (const float*)d_in[4];
  const float*         b1      = (const float*)d_in[5];
  const float*         W2      = (const float*)d_in[6];
  const float*         b2      = (const float*)d_in[7];
  const float*         Wa1     = (const float*)d_in[8];
  const float*         ba1     = (const float*)d_in[9];
  const float*         Wa2     = (const float*)d_in[10];
  const float*         ba2     = (const float*)d_in[11];
  const float*         Wl      = (const float*)d_in[12];
  const float*         bl      = (const float*)d_in[13];
  float*               out     = (float*)d_out;

  char* w = (char*)d_ws;
  size_t off = 0;
  __bf16* xws  = (__bf16*)(w + off); off += (size_t)kM * kN1 * sizeof(__bf16);   // 16 MB
  float*  whws = (float*) (w + off); off += (size_t)kM * kH  * sizeof(float);    // 48 MB
  __bf16* whbf = (__bf16*)(w + off); off += (size_t)kM * kH  * sizeof(__bf16);   // 24 MB
  float*  lgws = (float*) (w + off); off += (size_t)kM * sizeof(float);          // 64 KB
  __bf16* W1bf = (__bf16*)(w + off); off += (size_t)kN1 * kH * sizeof(__bf16);   // 0.75 MB
  __bf16* W2bf = (__bf16*)(w + off); off += (size_t)kH * kN1 * sizeof(__bf16);   // 0.75 MB
  __bf16* Wa1bf = (__bf16*)(w + off); off += (size_t)kNApad * kH * sizeof(__bf16);

  // Weight pre-conversion (tiny; once per launch)
  k_cvt_bf16<<<(kN1 * kH / 4 + 255) / 256, 256, 0, stream>>>(W1, W1bf, kN1 * kH / 4);
  k_cvt_bf16<<<(kH * kN1 / 4 + 255) / 256, 256, 0, stream>>>(W2, W2bf, kH * kN1 / 4);
  k_cvt_wa1<<<(kNApad * kH + 255) / 256, 256, 0, stream>>>(Wa1, Wa1bf);

  // GEMM1: (16384/64)*(512/64) = 2048 waves, 4 waves/block
  k_gemm1<<<(kM / 64) * (kN1 / 64) / 4, 128, 0, stream>>>(table, ids, W1bf, b1, xws);
  // GEMM2: (16384/64)*(768/64) = 3072 waves
  k_gemm2<<<(kM / 64) * (kH / 64) / 4, 128, 0, stream>>>(xws, W2bf, b2, ids, in_tab, whws, whbf);
  // attention scores: 1024 waves (16 rows each)
  k_attn_scores<<<(kM / 16) / 4, 128, 0, stream>>>(whbf, Wa1bf, ba1, Wa2, ba2, lgws);
  k_finalize<<<kB, 256, 0, stream>>>(lgws, whws, hiddens, Wl, bl, out);
}
